// SelfAttention_1082331759108
// MI455X (gfx1250) — matrix operand order repaired and checked by reading, not compiled, not run on hardware
//
#include <hip/hip_runtime.h>

// ---------------------------------------------------------------------------
// Causal self-attention for MI455X (gfx1250, wave32, WMMA 16x16x32 f16).
//   q = (x @ w_q) * 1/sqrt(128) ; k = x @ w_k ; vt = (x @ w_v)^T   (f16 in ws)
//   out = softmax(causal(q @ k^T)) @ v                              (f32)
// K and V^T tiles are staged into double-buffered LDS with
// GLOBAL_LOAD_ASYNC_TO_LDS_B128 (ASYNCcnt) so DMA overlaps WMMA+softmax.
// B-fragments are preloaded in register groups so WMMAs issue back-to-back.
// ---------------------------------------------------------------------------

typedef __attribute__((ext_vector_type(16))) _Float16 v16h;
typedef __attribute__((ext_vector_type(8)))  _Float16 v8h;
typedef __attribute__((ext_vector_type(8)))  float    v8f;
typedef __attribute__((ext_vector_type(4)))  float    v4f;

#define N_TOK  8192
#define D_IN   1024
#define D_HEAD 128

// ---- CDNA5 async global->LDS copy (16B per lane, ASYNCcnt-tracked) ---------
__device__ __forceinline__ unsigned lds_off(const void* p) {
  // generic shared address: low 32 bits are the LDS byte offset
  return (unsigned)(unsigned long long)p;
}
__device__ __forceinline__ void async_copy16(unsigned ldsOff, const void* g) {
  asm volatile("global_load_async_to_lds_b128 %0, %1, off"
               :: "v"(ldsOff), "v"(g) : "memory");
}
__device__ __forceinline__ void wait_async0() {
  asm volatile("s_wait_asynccnt 0" ::: "memory");
}

// ---- WMMA fragment loaders (layouts per CDNA5 ISA 7.12.2) ------------------
// A-matrix 16xK (f16): lane L holds row M = L&15.
//   lanes 0-15 : elems 0..7 -> K = k0+0..7   ; elems 8..15 -> K = k0+16..23
//   lanes16-31 : elems 0..7 -> K = k0+8..15  ; elems 8..15 -> K = k0+24..31
__device__ __forceinline__ v16h frag_a(const _Float16* base, int stride,
                                       int row0, int k0, int lane) {
  const _Float16* p = base + (row0 + (lane & 15)) * stride
                           + k0 + ((lane >> 4) & 1) * 8;
  v8h lo = *(const v8h*)p;        // K = f .. f+7
  v8h hi = *(const v8h*)(p + 16); // K = f+16 .. f+23
  v16h r;
#pragma unroll
  for (int i = 0; i < 8; ++i) { r[i] = lo[i]; r[i + 8] = hi[i]; }
  return r;
}

// B-matrix Kx16 (f16), stored as [n][k] (K contiguous):
//   lane L holds column N = L&15; lanes 0-15 K = k0+0..7,16..23, lanes 16-31 K = k0+8..15,24..31
__device__ __forceinline__ v16h frag_b(const _Float16* base, int stride,
                                       int col0, int k0, int lane) {
  const _Float16* p = base + (col0 + (lane & 15)) * stride
                           + k0 + ((lane >> 4) & 1) * 8;
  v8h lo = *(const v8h*)p;
  v8h hi = *(const v8h*)(p + 16);
  v16h r;
#pragma unroll
  for (int i = 0; i < 8; ++i) { r[i] = lo[i]; r[i + 8] = hi[i]; }
  return r;
}

// ---- Kernel 1: projection  C[8192x128] = f16(X[8192x1024] @ W[1024x128]) ---
// transposed==0: out[row][col] ; transposed==1: out[col][row]  (for V^T)
__launch_bounds__(128)
__global__ void qkv_proj_kernel(const float* __restrict__ x,
                                const float* __restrict__ w,
                                _Float16* __restrict__ outp,
                                float outScale, int transposed) {
  __shared__ __attribute__((aligned(16))) _Float16 Xs[64][72];   // 64 rows x 64 K (+pad)
  __shared__ __attribute__((aligned(16))) _Float16 Wt[128][72];  // [col][k] transposed

  const int tid  = threadIdx.x;
  const int lane = tid & 31;
  const int wave = tid >> 5;
  const int rowBase = blockIdx.x * 64;

  v8f acc[8];
#pragma unroll
  for (int nt = 0; nt < 8; ++nt)
#pragma unroll
    for (int i = 0; i < 8; ++i) acc[nt][i] = 0.0f;

  for (int k0 = 0; k0 < D_IN; k0 += 64) {
    __syncthreads();
    // stage X tile (f32 -> f16): 64 rows x 64 K
    for (int idx = tid; idx < 64 * 16; idx += 128) {
      int r = idx >> 4, c4 = (idx & 15) * 4;
      v4f xv = *(const v4f*)&x[(size_t)(rowBase + r) * D_IN + k0 + c4];
#pragma unroll
      for (int i = 0; i < 4; ++i) Xs[r][c4 + i] = (_Float16)xv[i];
    }
    // stage W tile transposed (f32 -> f16): 64 K x 128 cols
    for (int idx = tid; idx < 64 * 32; idx += 128) {
      int r = idx >> 5, c4 = (idx & 31) * 4;
      v4f wv = *(const v4f*)&w[(size_t)(k0 + r) * D_HEAD + c4];
#pragma unroll
      for (int i = 0; i < 4; ++i) Wt[c4 + i][r] = (_Float16)wv[i];
    }
    __syncthreads();

#pragma unroll
    for (int ks = 0; ks < 2; ++ks) {
      v16h a = frag_a(&Xs[wave * 16][0], 72, 0, ks * 32, lane);
      v16h bfr[8];
#pragma unroll
      for (int nt = 0; nt < 8; ++nt)       // batch the ds_loads ...
        bfr[nt] = frag_b(&Wt[0][0], 72, nt * 16, ks * 32, lane);
#pragma unroll
      for (int nt = 0; nt < 8; ++nt)       // ... then WMMAs back-to-back
        acc[nt] = __builtin_amdgcn_wmma_f32_16x16x32_f16(
            false, a, false, bfr[nt], (short)0, acc[nt], false, false);
    }
  }

  const int nCol = lane & 15;
  const int mOff = ((lane >> 4) & 1) * 8;
#pragma unroll
  for (int nt = 0; nt < 8; ++nt)
#pragma unroll
    for (int i = 0; i < 8; ++i) {
      int row = rowBase + wave * 16 + mOff + i;
      int col = nt * 16 + nCol;
      _Float16 val = (_Float16)(acc[nt][i] * outScale);
      if (transposed)
        outp[(size_t)col * N_TOK + row] = val;
      else
        outp[(size_t)row * D_HEAD + col] = val;
    }
}

// ---- Kernel 2: causal flash attention --------------------------------------
// Block = 4 waves x 16 query rows (BM=64); key blocks of BN=32, double-buffered.
__launch_bounds__(128)
__global__ void flash_attn_kernel(const _Float16* __restrict__ q,
                                  const _Float16* __restrict__ k,
                                  const _Float16* __restrict__ vt,  // [feat][tok]
                                  float* __restrict__ out) {
  __shared__ __attribute__((aligned(16))) _Float16 Ks[2][32][136];   // [key][feat]
  __shared__ __attribute__((aligned(16))) _Float16 Vt[2][128][40];   // [feat][key]
  __shared__ __attribute__((aligned(16))) _Float16 Ps[4][16][40];    // per-wave P

  const int tid  = threadIdx.x;
  const int lane = tid & 31;
  const int wave = tid >> 5;
  const int qBase = blockIdx.x * 64;
  const int qRow  = qBase + wave * 16;
  const int nCol  = lane & 15;
  const int mOff  = ((lane >> 4) & 1) * 8;

  // async-DMA one BN=32 key block (K row-major, V^T feature-major) into LDS
  auto stage = [&](int buf, int j0) {
    for (int t = tid; t < 32 * 16; t += 128) {        // K: 32 keys x 128 feats
      int r = t >> 4, c8 = (t & 15) * 8;
      async_copy16(lds_off(&Ks[buf][r][c8]),
                   &k[(size_t)(j0 + r) * D_HEAD + c8]);
    }
    for (int t = tid; t < 128 * 4; t += 128) {        // V^T: 128 feats x 32 keys
      int f = t >> 2, c8 = (t & 3) * 8;
      async_copy16(lds_off(&Vt[buf][f][c8]),
                   &vt[(size_t)f * N_TOK + j0 + c8]);
    }
  };

  const int nb = (qBase + 64) / 32;  // causal: key blocks for this query block
  stage(0, 0);

  // Q row-block (16 x 128) as 4 A-fragments (overlaps with async DMA)
  v16h qf[4];
#pragma unroll
  for (int kk = 0; kk < 4; ++kk)
    qf[kk] = frag_a(q, D_HEAD, qRow, kk * 32, lane);

  v8f o[8];
#pragma unroll
  for (int nt = 0; nt < 8; ++nt)
#pragma unroll
    for (int i = 0; i < 8; ++i) o[nt][i] = 0.0f;
  float mrow[8], lrow[8];
#pragma unroll
  for (int i = 0; i < 8; ++i) { mrow[i] = -__builtin_inff(); lrow[i] = 0.0f; }

  wait_async0();
  __syncthreads();

  for (int jb = 0; jb < nb; ++jb) {
    const int cur = jb & 1;
    const int j0  = jb * 32;
    if (jb + 1 < nb) stage(1 - cur, j0 + 32);   // prefetch next block via DMA

    // S(16x32) = Q @ K^T : preload 4 B-frags, then 4 back-to-back WMMAs
    v8f s[2];
#pragma unroll
    for (int nt = 0; nt < 2; ++nt) {
#pragma unroll
      for (int i = 0; i < 8; ++i) s[nt][i] = 0.0f;
      v16h bk[4];
#pragma unroll
      for (int kk = 0; kk < 4; ++kk)
        bk[kk] = frag_b(&Ks[cur][0][0], 136, nt * 16, kk * 32, lane);
#pragma unroll
      for (int kk = 0; kk < 4; ++kk)
        s[nt] = __builtin_amdgcn_wmma_f32_16x16x32_f16(
            false, qf[kk], false, bk[kk], (short)0, s[nt], false, false);
    }

    // causal mask + per-row block max
    float bmax[8];
#pragma unroll
    for (int i = 0; i < 8; ++i) bmax[i] = -__builtin_inff();
#pragma unroll
    for (int nt = 0; nt < 2; ++nt) {
      int gcol = j0 + nt * 16 + nCol;
#pragma unroll
      for (int i = 0; i < 8; ++i) {
        int grow = qRow + mOff + i;
        float sv = (grow >= gcol) ? s[nt][i] : -__builtin_inff();
        s[nt][i] = sv;
        bmax[i] = fmaxf(bmax[i], sv);
      }
    }
#pragma unroll
    for (int i = 0; i < 8; ++i) {
#pragma unroll
      for (int m = 1; m < 16; m <<= 1)
        bmax[i] = fmaxf(bmax[i], __shfl_xor(bmax[i], m, 32));
    }

    float scale[8];
#pragma unroll
    for (int i = 0; i < 8; ++i) {
      float newm = fmaxf(mrow[i], bmax[i]);
      scale[i] = __expf(mrow[i] - newm);  // first block: exp(-inf)=0
      mrow[i] = newm;
    }

    // P = exp(S - m): stage f16 through per-wave LDS (C-layout -> A-layout)
    float bsum[8];
#pragma unroll
    for (int i = 0; i < 8; ++i) bsum[i] = 0.0f;
#pragma unroll
    for (int nt = 0; nt < 2; ++nt) {
#pragma unroll
      for (int i = 0; i < 8; ++i) {
        float p = __expf(s[nt][i] - mrow[i]);
        bsum[i] += p;
        Ps[wave][mOff + i][nt * 16 + nCol] = (_Float16)p;
      }
    }
#pragma unroll
    for (int i = 0; i < 8; ++i) {
#pragma unroll
      for (int m = 1; m < 16; m <<= 1)
        bsum[i] += __shfl_xor(bsum[i], m, 32);
      lrow[i] = lrow[i] * scale[i] + bsum[i];
    }

    // O = O * scale + P @ V   (two groups of 4 preloaded B-frags)
#pragma unroll
    for (int nt = 0; nt < 8; ++nt)
#pragma unroll
      for (int i = 0; i < 8; ++i) o[nt][i] *= scale[i];

    v16h pf = frag_a(&Ps[wave][0][0], 40, 0, 0, lane);
#pragma unroll
    for (int g = 0; g < 2; ++g) {
      v16h bv[4];
#pragma unroll
      for (int nt = 0; nt < 4; ++nt)
        bv[nt] = frag_b(&Vt[cur][0][0], 40, (g * 4 + nt) * 16, 0, lane);
#pragma unroll
      for (int nt = 0; nt < 4; ++nt)
        o[g * 4 + nt] = __builtin_amdgcn_wmma_f32_16x16x32_f16(
            false, pf, false, bv[nt], (short)0, o[g * 4 + nt], false, false);
    }

    wait_async0();      // my async writes for the next buffer have landed
    __syncthreads();    // everyone done reading cur / writing next
  }

  // epilogue: normalize and store f32
#pragma unroll
  for (int i = 0; i < 8; ++i) {
    float inv = 1.0f / lrow[i];
    int grow = qRow + mOff + i;
#pragma unroll
    for (int nt = 0; nt < 8; ++nt)
      out[(size_t)grow * D_HEAD + nt * 16 + nCol] = o[nt][i] * inv;
  }
}

// ---------------------------------------------------------------------------
extern "C" void kernel_launch(void* const* d_in, const int* in_sizes, int n_in,
                              void* d_out, int out_size, void* d_ws, size_t ws_size,
                              hipStream_t stream) {
  const float* x  = (const float*)d_in[0];
  const float* wq = (const float*)d_in[1];
  const float* wk = (const float*)d_in[2];
  const float* wv = (const float*)d_in[3];

  _Float16* qb  = (_Float16*)d_ws;                     // 8192*128 f16 = 2 MB
  _Float16* kb  = qb + (size_t)N_TOK * D_HEAD;
  _Float16* vtb = kb + (size_t)N_TOK * D_HEAD;         // stored [feat][tok]
  float* out = (float*)d_out;

  const float qscale = 0.08838834764831843f;           // 1/sqrt(128)
  dim3 blk(128);
  qkv_proj_kernel<<<N_TOK / 64, blk, 0, stream>>>(x, wq, qb,  qscale, 0);
  qkv_proj_kernel<<<N_TOK / 64, blk, 0, stream>>>(x, wk, kb,  1.0f,   0);
  qkv_proj_kernel<<<N_TOK / 64, blk, 0, stream>>>(x, wv, vtb, 1.0f,   1);
  flash_attn_kernel<<<N_TOK / 64, blk, 0, stream>>>(qb, kb, vtb, out);
}
